// MultiHeadLatentAttention_12841952215128
// MI455X (gfx1250) — compile-verified, matrix operand order checked by reading, not run on hardware
//
#include <hip/hip_runtime.h>
#include <hip/hip_bf16.h>

// ---------------------------------------------------------------------------
// MLA prefill for gfx1250 (MI455X): bf16 WMMA everywhere, async-LDS staged
// GEMM weights, flash attention.
// ---------------------------------------------------------------------------

#define S_LEN 2048
#define DIM_  2048
#define H_    16
#define QLR_  512
#define KVLR_ 512
#define NOPE_ 128
#define ROPE_ 64
#define QKHD_ 192
#define VHD_  128
#define EPS_  1.1920929e-07f

typedef __bf16 bf16;
typedef __attribute__((ext_vector_type(16))) __bf16 v16bf;
typedef __attribute__((ext_vector_type(8)))  __bf16 v8bf;
typedef __attribute__((ext_vector_type(8)))  float  v8f;

__device__ __forceinline__ v8f zero_v8f() {
  v8f z;
#pragma unroll
  for (int i = 0; i < 8; ++i) z[i] = 0.0f;
  return z;
}

__device__ __forceinline__ v8f wmma_bf16(v16bf a, v16bf b, v8f c) {
  // D = A(16x32 bf16) * B(32x16 bf16) + C(16x16 f32)
  return __builtin_amdgcn_wmma_f32_16x16x32_bf16(
      /*neg_a=*/false, a, /*neg_b=*/false, b,
      /*c_mod=*/(short)0, c, /*reuse_a=*/false, /*reuse_b=*/false);
}

// Load a 16x32 bf16 fragment (A-style; B-style mirrors it) from a row-major
// [rows x ld] matrix. Per ISA 16-bit A layout: lane group g=lane/16 holds
// K in {8g..8g+7} U {16+8g..16+8g+7} -> two contiguous 16B loads per lane.
__device__ __forceinline__ v16bf load_frag(const bf16* __restrict__ base,
                                           int ld, int row0, int k0) {
  const int lane = threadIdx.x & 31;
  const int g = lane >> 4;
  const bf16* p = base + (size_t)(row0 + (lane & 15)) * ld + k0 + 8 * g;
  v8bf lo = *(const v8bf*)(p);
  v8bf hi = *(const v8bf*)(p + 16);
  v16bf f;
#pragma unroll
  for (int i = 0; i < 8; ++i) { f[i] = lo[i]; f[i + 8] = hi[i]; }
  return f;
}

// Async global->LDS copy of one 16B chunk per lane (gfx1250 ASYNCcnt path).
__device__ __forceinline__ void async_copy16(const bf16* gsrc, bf16* lds_dst) {
  unsigned lds_off = (unsigned)(uintptr_t)lds_dst;  // low 32 bits = LDS offset
  asm volatile("global_load_async_to_lds_b128 %0, %1, off"
               :: "v"(lds_off), "v"(gsrc) : "memory");
}

__device__ __forceinline__ void wait_asynccnt0() {
  asm volatile("s_wait_asynccnt 0x0" ::: "memory");
}

// ---------------------------------------------------------------------------
// fp32 -> bf16 conversion
// ---------------------------------------------------------------------------
__global__ void f32_to_bf16_kernel(const float* __restrict__ in,
                                   bf16* __restrict__ out, int n) {
  int i = blockIdx.x * blockDim.x + threadIdx.x;
  if (i < n) out[i] = (bf16)in[i];
}

// ---------------------------------------------------------------------------
// Y[M x N] = A[M x K](bf16) * W[N x K]^T(bf16) + bias[N], f32 output.
// One block owns a 128x64 output tile: 8 waves x (16 rows each), all sharing
// one 64-col weight strip. The 64x32 B tile for each k-step is staged into
// LDS with async global->LDS b128 copies (double buffered, one async
// instruction per wave per step), and read back as ds_load_b128 fragments.
// Requires: M%128==0, N%64==0, K%32==0.
// ---------------------------------------------------------------------------
__global__ __launch_bounds__(256) void gemm_bias_wmma(
    const bf16* __restrict__ A, const bf16* __restrict__ W,
    const float* __restrict__ bias, float* __restrict__ C,
    int M, int N, int K) {
  __shared__ __align__(16) bf16 bstage[2][64 * 32];  // 2 x 4KB ping-pong

  const int lane = threadIdx.x & 31;
  const int wave = threadIdx.x >> 5;
  const int bm = blockIdx.x % (M >> 7);
  const int n4 = blockIdx.x / (M >> 7);
  const int m0 = bm * 128 + wave * 16;  // this wave's 16 output rows

  // This thread's 16B chunk of the 64x32 B tile: 256 chunks == 256 threads.
  const int crow = threadIdx.x >> 2;          // 0..63  (row within N-strip)
  const int coff = (threadIdx.x & 3) * 8;     // 0/8/16/24 (bf16 elems)
  const bf16* wrow = W + (size_t)(n4 * 64 + crow) * K + coff;

  // prime buffer 0 with k=0
  async_copy16(wrow, &bstage[0][crow * 32 + coff]);

  v8f acc[4];
#pragma unroll
  for (int j = 0; j < 4; ++j) acc[j] = zero_v8f();

  for (int k = 0; k < K; k += 32) {
    const int cur = (k >> 5) & 1;
    wait_asynccnt0();      // this wave's copy into bstage[cur] landed
    __syncthreads();       // every wave's copy landed; prior reads of cur done

    if (k + 32 < K)        // overlap next tile copy with this step's WMMAs
      async_copy16(wrow + k + 32, &bstage[cur ^ 1][crow * 32 + coff]);

    v16bf a = load_frag(A, K, m0, k);
    if (k + 32 < K)
      __builtin_prefetch(A + (size_t)(m0 + (lane & 15)) * K + k + 32, 0, 1);
#pragma unroll
    for (int j = 0; j < 4; ++j) {
      v16bf b = load_frag(&bstage[cur][0], 32, j * 16, 0);  // ds_load_b128 x2
      acc[j] = wmma_bf16(a, b, acc[j]);
    }
    __syncthreads();       // all reads of bstage[cur] done before it is reused
  }

  const int g = lane >> 4, nl = lane & 15;
#pragma unroll
  for (int j = 0; j < 4; ++j) {
    int n = n4 * 64 + j * 16 + nl;
    float bb = bias[n];
#pragma unroll
    for (int v = 0; v < 8; ++v) {
      int m = m0 + v + 8 * g;  // C/D layout: row = vgpr + 8*(lane/16)
      C[(size_t)m * N + n] = acc[j][v] + bb;
    }
  }
}

// ---------------------------------------------------------------------------
// RMSNorm over `cols` leading columns of each row (row stride ld_in),
// bf16 output packed [rows x cols].
// ---------------------------------------------------------------------------
__global__ __launch_bounds__(256) void rmsnorm_bf16_kernel(
    const float* __restrict__ in, const float* __restrict__ w,
    bf16* __restrict__ out, int ld_in, int cols) {
  __shared__ float red[256];
  const int s = blockIdx.x;
  const float* row = in + (size_t)s * ld_in;
  float acc = 0.0f;
  for (int c = threadIdx.x; c < cols; c += 256) { float v = row[c]; acc += v * v; }
  red[threadIdx.x] = acc;
  __syncthreads();
  for (int o = 128; o > 0; o >>= 1) {
    if ((int)threadIdx.x < o) red[threadIdx.x] += red[threadIdx.x + o];
    __syncthreads();
  }
  const float r = rsqrtf(red[0] / (float)cols + EPS_);
  for (int c = threadIdx.x; c < cols; c += 256)
    out[(size_t)s * cols + c] = (bf16)(row[c] * r * w[c]);
}

// ---------------------------------------------------------------------------
// q_full f32 [S][H*192] -> RoPE on last 64 dims/head -> q_heads bf16 [H][S][192]
// ---------------------------------------------------------------------------
__global__ __launch_bounds__(256) void rope_q_kernel(
    const float* __restrict__ qf, const float* __restrict__ cosb,
    const float* __restrict__ sinb, bf16* __restrict__ qh) {
  const int s = blockIdx.x;
  for (int i = threadIdx.x; i < H_ * QKHD_; i += 256) {
    const int h = i / QKHD_, d = i % QKHD_;
    const float* qrow = qf + (size_t)s * (H_ * QKHD_) + h * QKHD_;
    float val;
    if (d < NOPE_) {
      val = qrow[d];
    } else {
      const int j = d - NOPE_, p = j >> 1;
      const float x0 = qrow[NOPE_ + 2 * p], x1 = qrow[NOPE_ + 2 * p + 1];
      const float c = cosb[s * (ROPE_ / 2) + p], sn = sinb[s * (ROPE_ / 2) + p];
      val = (j & 1) ? (x0 * sn + x1 * c) : (x0 * c - x1 * sn);
    }
    qh[((size_t)h * S_LEN + s) * QKHD_ + d] = (bf16)val;
  }
}

// ---------------------------------------------------------------------------
// kvb f32 [S][H*256] (k_nope | v), kva f32 [S][576] (latent | k_pe)
//  -> k_heads bf16 [H][S][192] (nope | roped k_pe broadcast)
//  -> v_t     bf16 [H][128][S]  (transposed so PV B-fragments load fast)
// ---------------------------------------------------------------------------
__global__ __launch_bounds__(256) void build_kv_kernel(
    const float* __restrict__ kvb, const float* __restrict__ kva,
    const float* __restrict__ cosb, const float* __restrict__ sinb,
    bf16* __restrict__ kh, bf16* __restrict__ vt) {
  const int s = blockIdx.x;
  for (int i = threadIdx.x; i < H_ * QKHD_; i += 256) {
    const int h = i / QKHD_, d = i % QKHD_;
    float val;
    if (d < NOPE_) {
      val = kvb[(size_t)s * (H_ * 256) + h * 256 + d];
    } else {
      const int j = d - NOPE_, p = j >> 1;
      const float x0 = kva[(size_t)s * 576 + KVLR_ + 2 * p];
      const float x1 = kva[(size_t)s * 576 + KVLR_ + 2 * p + 1];
      const float c = cosb[s * (ROPE_ / 2) + p], sn = sinb[s * (ROPE_ / 2) + p];
      val = (j & 1) ? (x0 * sn + x1 * c) : (x0 * c - x1 * sn);
    }
    kh[((size_t)h * S_LEN + s) * QKHD_ + d] = (bf16)val;
  }
  for (int i = threadIdx.x; i < H_ * VHD_; i += 256) {
    const int h = i / VHD_, d = i % VHD_;
    vt[((size_t)h * VHD_ + d) * S_LEN + s] =
        (bf16)kvb[(size_t)s * (H_ * 256) + h * 256 + NOPE_ + d];
  }
}

// ---------------------------------------------------------------------------
// Flash attention: grid (S/128, H), 8 waves/block, one 16-row q-tile per wave.
// Key tiles of 32: QK^T via 2x6 WMMAs, online softmax in C/D layout,
// P restaged through per-wave LDS, PV via 8 WMMAs against transposed V.
// Out: bf16 [S][H*128].
// ---------------------------------------------------------------------------
__global__ __launch_bounds__(256) void mla_flash_attn(
    const bf16* __restrict__ Qh, const bf16* __restrict__ Kh,
    const bf16* __restrict__ Vt, bf16* __restrict__ Out) {
  __shared__ __align__(16) bf16 plds[8][16 * 32];
  const float scale = 0.07216878364870323f;  // 192^-0.5
  const int lane = threadIdx.x & 31, wave = threadIdx.x >> 5;
  const int g = lane >> 4, nl = lane & 15;
  const int h = blockIdx.y;
  const int q0 = blockIdx.x * 128 + wave * 16;

  const bf16* Q = Qh + (size_t)h * S_LEN * QKHD_;
  const bf16* K = Kh + (size_t)h * S_LEN * QKHD_;
  const bf16* V = Vt + (size_t)h * VHD_ * S_LEN;

  v16bf qf[6];
#pragma unroll
  for (int c = 0; c < 6; ++c) qf[c] = load_frag(Q, QKHD_, q0, c * 32);

  v8f o[8];
#pragma unroll
  for (int nt = 0; nt < 8; ++nt) o[nt] = zero_v8f();
  float mrow[8], lrow[8];
#pragma unroll
  for (int v = 0; v < 8; ++v) { mrow[v] = -1e30f; lrow[v] = 0.0f; }

  for (int t0 = 0; t0 < q0 + 16; t0 += 32) {
    v8f st[2];
    st[0] = zero_v8f(); st[1] = zero_v8f();
#pragma unroll
    for (int half = 0; half < 2; ++half)
#pragma unroll
      for (int c = 0; c < 6; ++c) {
        v16bf kf = load_frag(K, QKHD_, t0 + half * 16, c * 32);
        st[half] = wmma_bf16(qf[c], kf, st[half]);
      }

#pragma unroll
    for (int v = 0; v < 8; ++v) {
      const int qm = q0 + v + 8 * g;  // this lane's query row for vgpr v
      float p0 = st[0][v] * scale;
      float p1 = st[1][v] * scale;
      if (t0 + nl > qm)      p0 = -1e30f;  // causal (== triu -1e9 mask)
      if (t0 + 16 + nl > qm) p1 = -1e30f;

      float tm = fmaxf(p0, p1);
      tm = fmaxf(tm, __shfl_xor(tm, 1, 32));
      tm = fmaxf(tm, __shfl_xor(tm, 2, 32));
      tm = fmaxf(tm, __shfl_xor(tm, 4, 32));
      tm = fmaxf(tm, __shfl_xor(tm, 8, 32));   // row max within 16-lane group
      const float mnew = fmaxf(mrow[v], tm);

      const float e0 = __expf(p0 - mnew);
      const float e1 = __expf(p1 - mnew);
      float rs = e0 + e1;
      rs += __shfl_xor(rs, 1, 32);
      rs += __shfl_xor(rs, 2, 32);
      rs += __shfl_xor(rs, 4, 32);
      rs += __shfl_xor(rs, 8, 32);

      const float co = __expf(mrow[v] - mnew);
      lrow[v] = lrow[v] * co + rs;
      mrow[v] = mnew;
#pragma unroll
      for (int nt = 0; nt < 8; ++nt) o[nt][v] *= co;

      const int m = v + 8 * g;  // P row in the 16x32 tile
      plds[wave][m * 32 + nl] = (bf16)e0;
      plds[wave][m * 32 + 16 + nl] = (bf16)e1;
    }

    asm volatile("s_wait_dscnt 0x0" ::: "memory");  // DS store -> DS load (same wave)
    v16bf pa = load_frag(&plds[wave][0], 32, 0, 0);

#pragma unroll
    for (int nt = 0; nt < 8; ++nt) {
      v16bf vf = load_frag(V, S_LEN, nt * 16, t0);  // B[k][n] = Vt[n][t0+k]
      o[nt] = wmma_bf16(pa, vf, o[nt]);
    }
  }

#pragma unroll
  for (int nt = 0; nt < 8; ++nt)
#pragma unroll
    for (int v = 0; v < 8; ++v) {
      const int m = q0 + v + 8 * g;
      Out[(size_t)m * (H_ * VHD_) + h * VHD_ + nt * 16 + nl] =
          (bf16)(o[nt][v] / lrow[v]);
    }
}

// ---------------------------------------------------------------------------
// Host launcher
// ---------------------------------------------------------------------------
extern "C" void kernel_launch(void* const* d_in, const int* in_sizes, int n_in,
                              void* d_out, int out_size, void* d_ws, size_t ws_size,
                              hipStream_t stream) {
  const float* x        = (const float*)d_in[0];
  const float* fcos     = (const float*)d_in[2];
  const float* fsin     = (const float*)d_in[3];
  const float* wq_a_w   = (const float*)d_in[5];
  const float* wq_a_b   = (const float*)d_in[6];
  const float* q_norm_w = (const float*)d_in[7];
  const float* wq_b_w   = (const float*)d_in[8];
  const float* wq_b_b   = (const float*)d_in[9];
  const float* wkv_a_w  = (const float*)d_in[10];
  const float* wkv_a_b  = (const float*)d_in[11];
  const float* kv_norm_w= (const float*)d_in[12];
  const float* wkv_b_w  = (const float*)d_in[13];
  const float* wkv_b_b  = (const float*)d_in[14];
  const float* wo_w     = (const float*)d_in[15];
  const float* wo_b     = (const float*)d_in[16];
  float* out = (float*)d_out;

  // Bump allocator over d_ws (256B aligned chunks).
  char* base = (char*)d_ws;
  size_t off = 0;
  auto alloc = [&](size_t bytes) -> char* {
    char* p = base + off;
    off = (off + bytes + 255) & ~(size_t)255;
    return p;
  };

  const size_t S = S_LEN;
  bf16* x_bf     = (bf16*)alloc(S * DIM_ * 2);
  bf16* wqa_bf   = (bf16*)alloc((size_t)QLR_ * DIM_ * 2);
  bf16* wqb_bf   = (bf16*)alloc((size_t)(H_ * QKHD_) * QLR_ * 2);
  bf16* wkva_bf  = (bf16*)alloc((size_t)(KVLR_ + ROPE_) * DIM_ * 2);
  bf16* wkvb_bf  = (bf16*)alloc((size_t)(H_ * 256) * KVLR_ * 2);
  bf16* wo_bf    = (bf16*)alloc((size_t)DIM_ * (H_ * VHD_) * 2);
  float* q_lat   = (float*)alloc(S * QLR_ * 4);
  float* kv_a    = (float*)alloc(S * (KVLR_ + ROPE_) * 4);
  bf16* q_n_bf   = (bf16*)alloc(S * QLR_ * 2);
  bf16* kv_n_bf  = (bf16*)alloc(S * KVLR_ * 2);
  float* q_full  = (float*)alloc(S * (size_t)(H_ * QKHD_) * 4);
  float* kvb     = (float*)alloc(S * (size_t)(H_ * 256) * 4);
  bf16* q_heads  = (bf16*)alloc((size_t)H_ * S * QKHD_ * 2);
  bf16* k_heads  = (bf16*)alloc((size_t)H_ * S * QKHD_ * 2);
  bf16* v_t      = (bf16*)alloc((size_t)H_ * VHD_ * S * 2);
  bf16* attn_bf  = (bf16*)alloc(S * (size_t)(H_ * VHD_) * 2);

  auto cvt = [&](const float* src, bf16* dst, int n) {
    f32_to_bf16_kernel<<<(n + 255) / 256, 256, 0, stream>>>(src, dst, n);
  };
  cvt(x,       x_bf,    S * DIM_);
  cvt(wq_a_w,  wqa_bf,  QLR_ * DIM_);
  cvt(wq_b_w,  wqb_bf,  (H_ * QKHD_) * QLR_);
  cvt(wkv_a_w, wkva_bf, (KVLR_ + ROPE_) * DIM_);
  cvt(wkv_b_w, wkvb_bf, (H_ * 256) * KVLR_);
  cvt(wo_w,    wo_bf,   DIM_ * (H_ * VHD_));

  auto gemm = [&](const bf16* A, const bf16* W, const float* b, float* C,
                  int M, int N, int K) {
    int blocks = (M >> 7) * (N >> 6);   // 128x64 output tile per block
    gemm_bias_wmma<<<blocks, 256, 0, stream>>>(A, W, b, C, M, N, K);
  };

  // q latent / kv latent projections
  gemm(x_bf, wqa_bf,  wq_a_b,  q_lat, S_LEN, QLR_, DIM_);
  gemm(x_bf, wkva_bf, wkv_a_b, kv_a,  S_LEN, KVLR_ + ROPE_, DIM_);

  // rmsnorms -> bf16 operands
  rmsnorm_bf16_kernel<<<S_LEN, 256, 0, stream>>>(q_lat, q_norm_w,  q_n_bf,  QLR_, QLR_);
  rmsnorm_bf16_kernel<<<S_LEN, 256, 0, stream>>>(kv_a,  kv_norm_w, kv_n_bf, KVLR_ + ROPE_, KVLR_);

  // up-projections
  gemm(q_n_bf,  wqb_bf,  wq_b_b,  q_full, S_LEN, H_ * QKHD_, QLR_);
  gemm(kv_n_bf, wkvb_bf, wkv_b_b, kvb,    S_LEN, H_ * 256,   KVLR_);

  // rope + head layout
  rope_q_kernel<<<S_LEN, 256, 0, stream>>>(q_full, fcos, fsin, q_heads);
  build_kv_kernel<<<S_LEN, 256, 0, stream>>>(kvb, kv_a, fcos, fsin, k_heads, v_t);

  // flash attention: grid (S/128, H)
  dim3 agrid(S_LEN / 128, H_);
  mla_flash_attn<<<agrid, 256, 0, stream>>>(q_heads, k_heads, v_t, attn_bf);

  // output projection straight into d_out (f32)
  gemm(attn_bf, wo_bf, wo_b, out, S_LEN, DIM_, H_ * VHD_);

  (void)in_sizes; (void)n_in; (void)out_size; (void)ws_size;
}
